// ChunkedCrossAttention_50525995270196
// MI455X (gfx1250) — compile-verified, matrix operand order checked by reading, not run on hardware
//
#include <hip/hip_runtime.h>
#include <hip/hip_bf16.h>
#include <stdint.h>

#define B_    4
#define SEQ   2048
#define DM    1024
#define NH    16
#define DK    64
#define CH    32
#define CL    64
#define JTOT  256                 // neighbors(2) * nlen(128)
#define MQ    (B_*CH*CL)          // 8192 query rows
#define ME    (B_*CH*2*128)       // 32768 e rows
#define VALID (SEQ - CL + 1)      // 1985 valid (unpadded) rows per batch

// --- TDM availability / arity detection (probe doc: 5-arg on ROCm7.2,
//     6-arg on amdgpu-toolchain which uniquely ships amd_gfx1250_TDM.h) ---
#if defined(__has_builtin)
#  if __has_builtin(__builtin_amdgcn_tensor_load_to_lds)
#    define HAVE_TDM 1
#  endif
#endif
#if defined(__has_include)
#  if __has_include(<hip/amd_detail/amd_gfx1250_TDM.h>)
#    define TDM_6ARG 1
#  endif
#endif

typedef __bf16 bf16x16 __attribute__((ext_vector_type(16)));
typedef float  f32x8   __attribute__((ext_vector_type(8)));
typedef unsigned int u32x4 __attribute__((ext_vector_type(4)));
typedef int          i32x4 __attribute__((ext_vector_type(4)));
typedef int          i32x8 __attribute__((ext_vector_type(8)));

union Frag { bf16x16 v; uint4 q[2]; unsigned short us[16]; unsigned int ui[8]; };
union Acc  { f32x8 v; float f[8]; };

__device__ __forceinline__ unsigned short f2bf(float x){
    unsigned int u = __float_as_uint(x);
    u += 0x7FFFu + ((u >> 16) & 1u);           // round-to-nearest-even
    return (unsigned short)(u >> 16);
}
__device__ __forceinline__ unsigned int pkbf(float a, float b){
    return (unsigned int)f2bf(a) | ((unsigned int)f2bf(b) << 16);
}

// ---------------------------------------------------------------------------
// 1) LayerNorm of h[:, 63:] -> hh bf16 (rows >= VALID zero-padded)
// ---------------------------------------------------------------------------
__global__ __launch_bounds__(256)
void ln_shift_kernel(const float* __restrict__ h, const float* __restrict__ gamma,
                     const float* __restrict__ beta, unsigned short* __restrict__ hh){
    const int row = blockIdx.x;               // 0..MQ-1
    const int b   = row >> 11;
    const int lp  = row & 2047;
    const int tid = threadIdx.x;              // 256 threads * 4 elems = 1024
    unsigned short* dst = hh + (size_t)row * DM;
    if (lp >= VALID){
        *(uint2*)(dst + tid*4) = make_uint2(0u, 0u);
        return;
    }
    const float* src = h + ((size_t)b*SEQ + lp + CL - 1)*DM;
    float4 x = *(const float4*)(src + tid*4);
    float s  = x.x + x.y + x.z + x.w;
    float ss = x.x*x.x + x.y*x.y + x.z*x.z + x.w*x.w;
    #pragma unroll
    for (int m = 16; m; m >>= 1){ s += __shfl_xor(s, m, 32); ss += __shfl_xor(ss, m, 32); }
    __shared__ float red_s[8], red_q[8];
    const int wv = tid >> 5, ln = tid & 31;
    if (ln == 0){ red_s[wv] = s; red_q[wv] = ss; }
    __syncthreads();
    if (wv == 0){
        float a  = (ln < 8) ? red_s[ln] : 0.f;
        float b2 = (ln < 8) ? red_q[ln] : 0.f;
        #pragma unroll
        for (int m = 4; m; m >>= 1){ a += __shfl_xor(a, m, 32); b2 += __shfl_xor(b2, m, 32); }
        if (ln == 0){ red_s[0] = a; red_q[0] = b2; }
    }
    __syncthreads();
    const float mu  = red_s[0] * (1.f/DM);
    const float var = red_q[0] * (1.f/DM) - mu*mu;
    const float rs  = rsqrtf(var + 1e-5f);
    float4 g  = *(const float4*)(gamma + tid*4);
    float4 be = *(const float4*)(beta  + tid*4);
    float o0 = (x.x-mu)*rs*g.x + be.x, o1 = (x.y-mu)*rs*g.y + be.y;
    float o2 = (x.z-mu)*rs*g.z + be.z, o3 = (x.w-mu)*rs*g.w + be.w;
    *(uint2*)(dst + tid*4) = make_uint2(pkbf(o0,o1), pkbf(o2,o3));
}

// ---------------------------------------------------------------------------
// 2) Transpose + f32->bf16 all four weights: W[K][N] -> Wt[N][K]
// ---------------------------------------------------------------------------
__global__ __launch_bounds__(256)
void transpose_cvt_kernel(const float* __restrict__ W0, const float* __restrict__ W1,
                          const float* __restrict__ W2, const float* __restrict__ W3,
                          unsigned short* __restrict__ T0, unsigned short* __restrict__ T1,
                          unsigned short* __restrict__ T2, unsigned short* __restrict__ T3){
    const float* W = (blockIdx.z==0)?W0:(blockIdx.z==1)?W1:(blockIdx.z==2)?W2:W3;
    unsigned short* T = (blockIdx.z==0)?T0:(blockIdx.z==1)?T1:(blockIdx.z==2)?T2:T3;
    __shared__ float tile[32][33];
    const int kx = blockIdx.y*32, nx = blockIdx.x*32;
    const int tx = threadIdx.x, ty = threadIdx.y;   // (32,8)
    #pragma unroll
    for (int i = 0; i < 32; i += 8)
        tile[ty+i][tx] = W[(size_t)(kx+ty+i)*DM + nx + tx];
    __syncthreads();
    #pragma unroll
    for (int i = 0; i < 32; i += 8)
        T[(size_t)(nx+ty+i)*DM + kx + tx] = f2bf(tile[tx][ty+i]);
}

// ---------------------------------------------------------------------------
// 3) WMMA GEMM: C_bf16[M][N] = A[M][K] @ Bt[N][K]^T + bias
//    A_F32=1 -> A is f32 (e), else bf16. One wave -> 32x64 tile
//    (2 A-fragments x 4 B-fragments -> 8 WMMA per K-step, B reused x2).
// ---------------------------------------------------------------------------
template<int A_F32>
__global__ __launch_bounds__(256)
void gemm_bias_bf16(const void* __restrict__ Ap, const unsigned short* __restrict__ Bt,
                    const float* __restrict__ bias, unsigned short* __restrict__ C,
                    int M, int N, int K){
    const int wave = (blockIdx.x * blockDim.x + threadIdx.x) >> 5;
    const int lane = threadIdx.x & 31;
    const int tn   = N >> 6;
    const int tm   = wave / tn;                 // 32-row tile
    const int nb   = (wave % tn) * 64;
    const int l16  = lane & 15;
    const int h8   = (lane >> 4) * 8;
    const int mrow0 = tm*32 + l16;
    const int mrow1 = mrow0 + 16;
    const unsigned short* a16 = (const unsigned short*)Ap;
    const float*          a32 = (const float*)Ap;

    Acc acc[2][4];
    #pragma unroll
    for (int u = 0; u < 2; u++)
        #pragma unroll
        for (int t = 0; t < 4; t++) acc[u][t].v = {};

    for (int k = 0; k < K; k += 32){
        Frag A0, A1;
        if (A_F32){
            const float* p0 = a32 + (size_t)mrow0*K + k + h8;
            const float* p1 = a32 + (size_t)mrow1*K + k + h8;
            float4 f0 = *(const float4*)(p0);
            float4 f1 = *(const float4*)(p0 + 4);
            float4 f2 = *(const float4*)(p0 + 16);
            float4 f3 = *(const float4*)(p0 + 20);
            A0.ui[0]=pkbf(f0.x,f0.y); A0.ui[1]=pkbf(f0.z,f0.w);
            A0.ui[2]=pkbf(f1.x,f1.y); A0.ui[3]=pkbf(f1.z,f1.w);
            A0.ui[4]=pkbf(f2.x,f2.y); A0.ui[5]=pkbf(f2.z,f2.w);
            A0.ui[6]=pkbf(f3.x,f3.y); A0.ui[7]=pkbf(f3.z,f3.w);
            f0 = *(const float4*)(p1);
            f1 = *(const float4*)(p1 + 4);
            f2 = *(const float4*)(p1 + 16);
            f3 = *(const float4*)(p1 + 20);
            A1.ui[0]=pkbf(f0.x,f0.y); A1.ui[1]=pkbf(f0.z,f0.w);
            A1.ui[2]=pkbf(f1.x,f1.y); A1.ui[3]=pkbf(f1.z,f1.w);
            A1.ui[4]=pkbf(f2.x,f2.y); A1.ui[5]=pkbf(f2.z,f2.w);
            A1.ui[6]=pkbf(f3.x,f3.y); A1.ui[7]=pkbf(f3.z,f3.w);
            if (k + 64 < K){ __builtin_prefetch(p0 + 64, 0, 0); __builtin_prefetch(p1 + 64, 0, 0); }
        } else {
            const unsigned short* p0 = a16 + (size_t)mrow0*K + k + h8;
            const unsigned short* p1 = a16 + (size_t)mrow1*K + k + h8;
            A0.q[0] = *(const uint4*)(p0); A0.q[1] = *(const uint4*)(p0 + 16);
            A1.q[0] = *(const uint4*)(p1); A1.q[1] = *(const uint4*)(p1 + 16);
            if (k + 64 < K){ __builtin_prefetch(p0 + 64, 0, 0); __builtin_prefetch(p1 + 64, 0, 0); }
        }
        #pragma unroll
        for (int t = 0; t < 4; t++){
            const unsigned short* bp = Bt + (size_t)(nb + t*16 + l16)*K + k + h8;
            Frag Bf; Bf.q[0] = *(const uint4*)(bp); Bf.q[1] = *(const uint4*)(bp + 16);
            acc[0][t].v = __builtin_amdgcn_wmma_f32_16x16x32_bf16(
                false, A0.v, false, Bf.v, (short)0, acc[0][t].v, false, false);
            acc[1][t].v = __builtin_amdgcn_wmma_f32_16x16x32_bf16(
                false, A1.v, false, Bf.v, (short)0, acc[1][t].v, false, false);
        }
    }
    #pragma unroll
    for (int u = 0; u < 2; u++){
        const int rbase = tm*32 + u*16 + h8;
        #pragma unroll
        for (int t = 0; t < 4; t++){
            const int col = nb + t*16 + l16;
            const float bvv = bias[col];
            #pragma unroll
            for (int r = 0; r < 8; r++)
                C[(size_t)(rbase + r)*N + col] = f2bf(acc[u][t].f[r] + bvv);
        }
    }
}

// ---------------------------------------------------------------------------
// 4) Attention: one workgroup (4 waves) per (b*32+c, head)
//    K tile staged to LDS via Tensor Data Mover (TDM) when available.
//    Dynamic LDS: K[256][64] | Vt[64][256] | P[4][16][256]  (96 KB)
// ---------------------------------------------------------------------------
#define SMEM_USHORTS (JTOT*DK + DK*JTOT + 4*16*JTOT)
#define SMEM_BYTES   (SMEM_USHORTS*2)

__global__ __launch_bounds__(128)
void attn_kernel(const unsigned short* __restrict__ Qb, const unsigned short* __restrict__ Kb,
                 const unsigned short* __restrict__ Vb, unsigned short* __restrict__ Ab){
    const int head = blockIdx.x & 15;
    const int bc   = blockIdx.x >> 4;       // b*32+c
    const int tid  = threadIdx.x;
    const int w    = tid >> 5;              // wave 0..3 -> i rows [16w,16w+16)
    const int lane = tid & 31;
    const int l16  = lane & 15;
    const int h8   = (lane >> 4) * 8;

    extern __shared__ __align__(16) unsigned short smem[];
    unsigned short (*Klds)[DK]      = (unsigned short (*)[DK])(smem);
    unsigned short (*Vt)[JTOT]      = (unsigned short (*)[JTOT])(smem + JTOT*DK);
    unsigned short (*Pl)[16][JTOT]  = (unsigned short (*)[16][JTOT])(smem + 2*JTOT*DK);

    const unsigned short* ksrc = Kb + (size_t)bc*JTOT*DM + head*DK;
    const unsigned short* vsrc = Vb + (size_t)bc*JTOT*DM + head*DK;

#if defined(HAVE_TDM)
    if (w == 0){
        // Build D# descriptor: 2D tile 64(x) x 256(y), 2-byte elems,
        // tensor row stride 1024 elems, dest = Klds (row-major).
        const unsigned long long ga = (unsigned long long)(uintptr_t)ksrc;
        const unsigned int       la = (unsigned int)(uintptr_t)&Klds[0][0];
        u32x4 g0; i32x8 g1; i32x4 g2 = {0,0,0,0}, g3 = {0,0,0,0};
        g0[0] = 1u;                                             // count=1
        g0[1] = la;                                             // lds_addr
        g0[2] = (unsigned int)(ga & 0xFFFFFFFFu);               // global_addr lo
        g0[3] = (unsigned int)((ga >> 32) & 0x1FFFFFFu) | 0x80000000u; // hi | type=2
        g1[0] = (1 << 16);                                      // data_size=2B
        g1[1] = (int)(DK   << 16);                              // tensor_dim0=64
        g1[2] = (int)(JTOT << 16);                              // tensor_dim1=256
        g1[3] = (int)(DK   << 16);                              // tile_dim0=64
        g1[4] = JTOT;                                           // tile_dim1=256
        g1[5] = DM;                                             // dim0_stride=1024
        g1[6] = 0; g1[7] = 0;
#if defined(TDM_6ARG)
        i32x8 g4 = {0,0,0,0,0,0,0,0};
        __builtin_amdgcn_tensor_load_to_lds(g0, g1, g2, g3, g4, 0);
#else
        __builtin_amdgcn_tensor_load_to_lds(g0, g1, g2, g3, 0);
#endif
        __builtin_amdgcn_s_wait_tensorcnt(0);
    }
#else
    for (int it = tid; it < JTOT*8; it += 128){
        const int j = it >> 3, dg = (it & 7) * 8;
        *(uint4*)&Klds[j][dg] = *(const uint4*)(ksrc + (size_t)j*DM + dg);
    }
#endif

    // stage V tile transposed into LDS (Bt layout for O = P @ V)
    for (int it = tid; it < JTOT*8; it += 128){
        const int j = it >> 3, dg = (it & 7) * 8;
        unsigned short tmp[8];
        *(uint4*)tmp = *(const uint4*)(vsrc + (size_t)j*DM + dg);
        #pragma unroll
        for (int x = 0; x < 8; x++) Vt[dg + x][j] = tmp[x];
    }
    __syncthreads();

    // S = Q @ K^T  (16 x 256 per wave); K tile in LDS is already Bt layout
    Acc S[16];
    #pragma unroll
    for (int jt = 0; jt < 16; jt++) S[jt].v = {};
    const unsigned short* qrow = Qb + (size_t)(bc*CL + w*16 + l16)*DM + head*DK;
    #pragma unroll
    for (int kk = 0; kk < DK; kk += 32){
        Frag A;
        A.q[0] = *(const uint4*)(qrow + kk + h8);
        A.q[1] = *(const uint4*)(qrow + kk + 16 + h8);
        #pragma unroll
        for (int jt = 0; jt < 16; jt++){
            Frag Bf;
            Bf.q[0] = *(const uint4*)&Klds[jt*16 + l16][kk + h8];
            Bf.q[1] = *(const uint4*)&Klds[jt*16 + l16][kk + 16 + h8];
            S[jt].v = __builtin_amdgcn_wmma_f32_16x16x32_bf16(
                false, A.v, false, Bf.v, (short)0, S[jt].v, false, false);
        }
    }

    // softmax over the 256-wide row; row m = r + h8
    const float scale = 0.125f;               // 1/sqrt(64)
    #pragma unroll
    for (int r = 0; r < 8; r++){
        float mx = -3.0e38f;
        #pragma unroll
        for (int jt = 0; jt < 16; jt++) mx = fmaxf(mx, S[jt].f[r]);
        #pragma unroll
        for (int m = 8; m; m >>= 1) mx = fmaxf(mx, __shfl_xor(mx, m, 32));
        mx *= scale;
        float sum = 0.f;
        #pragma unroll
        for (int jt = 0; jt < 16; jt++){
            const float p = __expf(S[jt].f[r]*scale - mx);
            S[jt].f[r] = p; sum += p;
        }
        #pragma unroll
        for (int m = 8; m; m >>= 1) sum += __shfl_xor(sum, m, 32);
        const float inv = 1.f / sum;
        #pragma unroll
        for (int jt = 0; jt < 16; jt++) S[jt].f[r] *= inv;
    }

    // scatter P (bf16) to LDS in row-major A-source layout
    #pragma unroll
    for (int jt = 0; jt < 16; jt++)
        #pragma unroll
        for (int r = 0; r < 8; r++)
            Pl[w][r + h8][jt*16 + l16] = f2bf(S[jt].f[r]);
    __syncthreads();

    // O = P @ V : K-dim = 256, A from Pl, B from Vt
    Acc O[4];
    #pragma unroll
    for (int t = 0; t < 4; t++) O[t].v = {};
    #pragma unroll
    for (int kk = 0; kk < JTOT; kk += 32){
        Frag A;
        A.q[0] = *(const uint4*)&Pl[w][l16][kk + h8];
        A.q[1] = *(const uint4*)&Pl[w][l16][kk + 16 + h8];
        #pragma unroll
        for (int t = 0; t < 4; t++){
            Frag Bf;
            Bf.q[0] = *(const uint4*)&Vt[t*16 + l16][kk + h8];
            Bf.q[1] = *(const uint4*)&Vt[t*16 + l16][kk + 16 + h8];
            O[t].v = __builtin_amdgcn_wmma_f32_16x16x32_bf16(
                false, A.v, false, Bf.v, (short)0, O[t].v, false, false);
        }
    }
    unsigned short* ob = Ab + (size_t)(bc*CL + w*16)*DM + head*DK;
    #pragma unroll
    for (int t = 0; t < 4; t++)
        #pragma unroll
        for (int r = 0; r < 8; r++)
            ob[(size_t)(r + h8)*DM + t*16 + l16] = f2bf(O[t].f[r]);
}

// ---------------------------------------------------------------------------
// 5) Output GEMM + bias + shift(+63) + residual, f32 store
// ---------------------------------------------------------------------------
__global__ __launch_bounds__(256)
void gemm_out_residual(const unsigned short* __restrict__ A, const unsigned short* __restrict__ WoT,
                       const float* __restrict__ bo, const float* __restrict__ hres,
                       float* __restrict__ out){
    const int wave = (blockIdx.x * blockDim.x + threadIdx.x) >> 5;
    const int lane = threadIdx.x & 31;
    const int tn   = DM >> 6;
    const int tm   = wave / tn;
    const int nb   = (wave % tn) * 64;
    const int l16  = lane & 15;
    const int h8   = (lane >> 4) * 8;
    const int mrow = tm*16 + l16;

    Acc acc[4];
    #pragma unroll
    for (int t = 0; t < 4; t++) acc[t].v = {};
    for (int k = 0; k < DM; k += 32){
        Frag Af;
        const unsigned short* p = A + (size_t)mrow*DM + k + h8;
        Af.q[0] = *(const uint4*)(p);
        Af.q[1] = *(const uint4*)(p + 16);
        if (k + 64 < DM) __builtin_prefetch(p + 64, 0, 0);
        #pragma unroll
        for (int t = 0; t < 4; t++){
            const unsigned short* bp = WoT + (size_t)(nb + t*16 + l16)*DM + k + h8;
            Frag Bf; Bf.q[0] = *(const uint4*)(bp); Bf.q[1] = *(const uint4*)(bp + 16);
            acc[t].v = __builtin_amdgcn_wmma_f32_16x16x32_bf16(
                false, Af.v, false, Bf.v, (short)0, acc[t].v, false, false);
        }
    }
    #pragma unroll
    for (int t = 0; t < 4; t++){
        const int col = nb + t*16 + l16;
        const float bvv = bo[col];
        #pragma unroll
        for (int r = 0; r < 8; r++){
            const int p  = tm*16 + h8 + r;
            const int b  = p >> 11;
            const int lp = p & 2047;
            if (lp < VALID){
                const size_t o = ((size_t)b*SEQ + lp + CL - 1)*DM + col;
                out[o] = hres[o] + acc[t].f[r] + bvv;
            }
        }
    }
}

// 6) rows t < 63: out = h (pure residual, attention contribution is zero-pad)
__global__ __launch_bounds__(256)
void prefix_copy(const float* __restrict__ h, float* __restrict__ out){
    const int i = blockIdx.x*blockDim.x + threadIdx.x;     // float4 index
    const int per_b = (CL-1)*(DM/4);
    if (i < B_*per_b){
        const int b = i / per_b, rem = i % per_b;
        const size_t o = (size_t)b*SEQ*(DM/4) + rem;
        ((float4*)out)[o] = ((const float4*)h)[o];
    }
}

// ---------------------------------------------------------------------------
extern "C" void kernel_launch(void* const* d_in, const int* in_sizes, int n_in,
                              void* d_out, int out_size, void* d_ws, size_t ws_size,
                              hipStream_t stream){
    (void)in_sizes; (void)n_in; (void)out_size; (void)ws_size;
    const float* h     = (const float*)d_in[0];
    const float* e     = (const float*)d_in[1];
    const float* Wq    = (const float*)d_in[2];
    const float* bq    = (const float*)d_in[3];
    const float* Wk    = (const float*)d_in[4];
    const float* bk    = (const float*)d_in[5];
    const float* Wv    = (const float*)d_in[6];
    const float* bv    = (const float*)d_in[7];
    const float* Wo    = (const float*)d_in[8];
    const float* bo    = (const float*)d_in[9];
    const float* gamma = (const float*)d_in[10];
    const float* beta  = (const float*)d_in[11];
    float* out = (float*)d_out;

    char* ws = (char*)d_ws;
    size_t off = 0;
    auto alloc = [&](size_t bytes)->void*{
        void* p = ws + off; off += (bytes + 255) & ~(size_t)255; return p;
    };
    unsigned short* hh  = (unsigned short*)alloc((size_t)MQ*DM*2);
    unsigned short* WqT = (unsigned short*)alloc((size_t)DM*DM*2);
    unsigned short* WkT = (unsigned short*)alloc((size_t)DM*DM*2);
    unsigned short* WvT = (unsigned short*)alloc((size_t)DM*DM*2);
    unsigned short* WoT = (unsigned short*)alloc((size_t)DM*DM*2);
    unsigned short* Qb  = (unsigned short*)alloc((size_t)MQ*DM*2);
    unsigned short* Kb  = (unsigned short*)alloc((size_t)ME*DM*2);
    unsigned short* Vb  = (unsigned short*)alloc((size_t)ME*DM*2);
    unsigned short* Ab  = (unsigned short*)alloc((size_t)MQ*DM*2);

    ln_shift_kernel<<<MQ, 256, 0, stream>>>(h, gamma, beta, hh);
    transpose_cvt_kernel<<<dim3(DM/32, DM/32, 4), dim3(32, 8), 0, stream>>>(
        Wq, Wk, Wv, Wo, WqT, WkT, WvT, WoT);

    const int wavesQ = (MQ/32)*(DM/64);   // 4096
    const int wavesE = (ME/32)*(DM/64);   // 16384
    gemm_bias_bf16<0><<<wavesQ/8, 256, 0, stream>>>(hh, WqT, bq, Qb, MQ, DM, DM);
    gemm_bias_bf16<1><<<wavesE/8, 256, 0, stream>>>(e,  WkT, bk, Kb, ME, DM, DM);
    gemm_bias_bf16<1><<<wavesE/8, 256, 0, stream>>>(e,  WvT, bv, Vb, ME, DM, DM);

    attn_kernel<<<B_*CH*NH, 128, SMEM_BYTES, stream>>>(Qb, Kb, Vb, Ab);

    prefix_copy<<<(B_*(CL-1)*(DM/4) + 255)/256, 256, 0, stream>>>(h, out);

    const int wavesO = (MQ/16)*(DM/64);   // 8192
    gemm_out_residual<<<wavesO/8, 256, 0, stream>>>(Ab, WoT, bo, h, out);
}
